// EnhancedWikiGraphSAGE_44796508897843
// MI455X (gfx1250) — matrix operand order repaired and checked by reading, not compiled, not run on hardware
//
#include <hip/hip_runtime.h>
#include <hip/hip_bf16.h>

// ---------------------------------------------------------------------------
// EnhancedWikiGraphSAGE on MI455X (gfx1250, wave32, WMMA)
// All dense GEMMs via v_wmma_f32_16x16x32_f16 (f16 in, f32 acc).
// ---------------------------------------------------------------------------

typedef _Float16 h16;
typedef _Float16 v8h  __attribute__((ext_vector_type(8)));
typedef _Float16 v16h __attribute__((ext_vector_type(16)));
typedef float    v8f  __attribute__((ext_vector_type(8)));

#define N_NODES 50000
#define N_EDGES 800000
#define IN_DIM  300
#define IN_PAD  320
#define HID     128
#define N_LAYERS 4
#define MULTI   ((N_LAYERS + 1) * HID)   // 640

// ------------------------------- small kernels ------------------------------

__global__ void k_fill_f32(float* __restrict__ p, int n, float v) {
    int i = blockIdx.x * blockDim.x + threadIdx.x;
    if (i < n) p[i] = v;
}

__global__ void k_deg(const int* __restrict__ dst, float* __restrict__ deg) {
    int e = blockIdx.x * blockDim.x + threadIdx.x;
    if (e < N_EDGES) atomicAdd(&deg[dst[e]], 1.0f);
}

__global__ void k_invdeg(float* __restrict__ deg) {
    int i = blockIdx.x * blockDim.x + threadIdx.x;
    if (i < N_NODES) deg[i] = 1.0f / fmaxf(deg[i], 1.0f);
}

// x [N, IN_DIM] f32  ->  x16 [N, IN_PAD] f16 (zero padded K tail)
__global__ void k_cvt_x(const float* __restrict__ x, h16* __restrict__ x16) {
    int i = blockIdx.x * blockDim.x + threadIdx.x;
    int total = N_NODES * IN_PAD;
    if (i >= total) return;
    int node = i / IN_PAD;
    int j    = i - node * IN_PAD;
    float v  = (j < IN_DIM) ? x[(size_t)node * IN_DIM + j] : 0.0f;
    x16[i] = (h16)v;
}

// W [K, HID] f32 -> Wt [HID, Kp] f16 (transposed, zero-padded K)
__global__ void k_wt_cvt(const float* __restrict__ W, h16* __restrict__ Wt,
                         int K, int Kp) {
    int i = blockIdx.x * blockDim.x + threadIdx.x;
    if (i >= HID * Kp) return;
    int n = i / Kp;
    int k = i - n * Kp;
    float v = (k < K) ? W[(size_t)k * HID + n] : 0.0f;
    Wt[i] = (h16)v;
}

// scatter-add: agg[dst] += h[src]   (4 floats per thread, L2-resident atomics)
__global__ void k_scatter(const int* __restrict__ src, const int* __restrict__ dst,
                          const float* __restrict__ h, float* __restrict__ agg) {
    int gid = blockIdx.x * blockDim.x + threadIdx.x;
    if (gid >= N_EDGES * (HID / 4)) return;
    int e = gid >> 5;              // HID/4 == 32 chunks per edge
    int c = (gid & 31) * 4;
    int s = src[e], d = dst[e];
    const float* hp = h   + (size_t)s * HID + c;
    float*       ap = agg + (size_t)d * HID + c;
    atomicAdd(ap + 0, hp[0]);
    atomicAdd(ap + 1, hp[1]);
    atomicAdd(ap + 2, hp[2]);
    atomicAdd(ap + 3, hp[3]);
}

// a16 = f16(agg * inv_deg[node])
__global__ void k_scale_cvt(const float* __restrict__ agg, const float* __restrict__ inv,
                            h16* __restrict__ a16) {
    int i = blockIdx.x * blockDim.x + threadIdx.x;
    if (i >= N_NODES * HID) return;
    int node = i >> 7;
    a16[i] = (h16)(agg[i] * inv[node]);
}

// LayerNorm + optional skip + ReLU. One wave per node (4 feats/lane).
__global__ __launch_bounds__(256)
void k_ln(const float* __restrict__ hn, float* __restrict__ h,
          h16* __restrict__ rep, int ldrep,
          const float* __restrict__ g, const float* __restrict__ b, int skip) {
    int node = blockIdx.x * blockDim.y + threadIdx.y;
    if (node >= N_NODES) return;
    int lane = threadIdx.x;
    const float* p = hn + (size_t)node * HID + lane * 4;
    float v0 = p[0], v1 = p[1], v2 = p[2], v3 = p[3];
    float s  = v0 + v1 + v2 + v3;
    float sq = v0 * v0 + v1 * v1 + v2 * v2 + v3 * v3;
    #pragma unroll
    for (int m = 16; m >= 1; m >>= 1) {
        s  += __shfl_xor(s,  m, 32);
        sq += __shfl_xor(sq, m, 32);
    }
    float mu  = s * (1.0f / HID);
    float var = sq * (1.0f / HID) - mu * mu;
    float rs  = rsqrtf(var + 1e-5f);

    float* hp = h   + (size_t)node * HID   + lane * 4;
    h16*   rp = rep + (size_t)node * ldrep + lane * 4;
    float vv[4] = {v0, v1, v2, v3};
    float ys[4];
    #pragma unroll
    for (int q = 0; q < 4; ++q) {
        int f = lane * 4 + q;
        float y = (vv[q] - mu) * rs * g[f] + b[f];
        if (skip) y += hp[q];
        ys[q] = fmaxf(y, 0.0f);
    }
    #pragma unroll
    for (int q = 0; q < 4; ++q) { hp[q] = ys[q]; rp[q] = (h16)ys[q]; }
}

// ------------------------------- WMMA GEMM ----------------------------------
// C[M,128] = relu?( A0[M,K0] @ B0^T + (A1[M,K1] @ B1^T) + bias )
// A row-major f16 with stride lda; B stored transposed: Bt[n][k], stride K.
// grid.x = ceil(M/16) row tiles; block = (32, 8): wave ty handles col tile ty.
__global__ __launch_bounds__(256)
void k_gemm(const h16* __restrict__ A0, int lda0, const h16* __restrict__ B0, int K0,
            const h16* __restrict__ A1, int lda1, const h16* __restrict__ B1, int K1,
            const float* __restrict__ bias,
            float* __restrict__ outF, h16* __restrict__ outH, int ldh,
            int M, int relu) {
    const int lane = threadIdx.x;          // 0..31
    const int lo   = lane & 15;
    const int hi   = lane >> 4;            // 0 or 1
    const int n0   = threadIdx.y * 16;     // column tile
    const int m0   = blockIdx.x * 16;      // row tile
    const int rowA = (m0 + lo < M) ? (m0 + lo) : (M - 1);  // clamp, no divergence

    v8f acc;
    float bv = bias[n0 + lo];
    #pragma unroll
    for (int r = 0; r < 8; ++r) acc[r] = bv;

    // ---- first operand pair ----
    {
        const h16* ar = A0 + (size_t)rowA * lda0;
        const h16* br = B0 + (size_t)(n0 + lo) * K0;
        for (int k0 = 0; k0 < K0; k0 += 32) {
            v16h a, b;
            // A fragment: lanes 0-15 hold K=[k0..k0+7],[k0+16..k0+23]; lanes 16-31 shifted +8
            ((v8h*)&a)[0] = *(const v8h*)(ar + k0 + hi * 8);
            ((v8h*)&a)[1] = *(const v8h*)(ar + k0 + 16 + hi * 8);
            // B fragment: same K layout as A (lanes 0-15 K=[k0..k0+7],[k0+16..k0+23]; lanes 16-31 shifted +8)
            ((v8h*)&b)[0] = *(const v8h*)(br + k0 + hi * 8);
            ((v8h*)&b)[1] = *(const v8h*)(br + k0 + 16 + hi * 8);
            acc = __builtin_amdgcn_wmma_f32_16x16x32_f16(
                false, a, false, b, (short)0, acc, false, false);
        }
    }
    // ---- optional second operand pair (SAGE: agg@lin_l + h@lin_r) ----
    if (K1 > 0) {
        const h16* ar = A1 + (size_t)rowA * lda1;
        const h16* br = B1 + (size_t)(n0 + lo) * K1;
        for (int k0 = 0; k0 < K1; k0 += 32) {
            v16h a, b;
            ((v8h*)&a)[0] = *(const v8h*)(ar + k0 + hi * 8);
            ((v8h*)&a)[1] = *(const v8h*)(ar + k0 + 16 + hi * 8);
            ((v8h*)&b)[0] = *(const v8h*)(br + k0 + hi * 8);
            ((v8h*)&b)[1] = *(const v8h*)(br + k0 + 16 + hi * 8);
            acc = __builtin_amdgcn_wmma_f32_16x16x32_f16(
                false, a, false, b, (short)0, acc, false, false);
        }
    }

    // ---- epilogue: C/D layout: VGPR r -> M = r + hi*8, N = lane%16 ----
    #pragma unroll
    for (int r = 0; r < 8; ++r) {
        int row = m0 + hi * 8 + r;
        if (row < M) {
            float v = acc[r];
            if (relu) v = fmaxf(v, 0.0f);
            int col = n0 + lo;
            if (outF) outF[(size_t)row * HID + col] = v;
            if (outH) outH[(size_t)row * ldh + col] = (h16)v;
        }
    }
}

// ------------------------------- host side ----------------------------------

extern "C" void kernel_launch(void* const* d_in, const int* in_sizes, int n_in,
                              void* d_out, int out_size, void* d_ws, size_t ws_size,
                              hipStream_t stream) {
    const float* x        = (const float*)d_in[0];
    const int*   eidx     = (const int*)  d_in[1];
    const float* emb_W    = (const float*)d_in[2];
    const float* emb_b    = (const float*)d_in[3];
    const float* lin_l_W  = (const float*)d_in[4];
    const float* lin_l_b  = (const float*)d_in[5];
    const float* lin_r_W  = (const float*)d_in[6];
    const float* ln_g     = (const float*)d_in[7];
    const float* ln_b     = (const float*)d_in[8];
    const float* fus_W1   = (const float*)d_in[9];
    const float* fus_b1   = (const float*)d_in[10];
    const float* fus_W2   = (const float*)d_in[11];
    const float* fus_b2   = (const float*)d_in[12];
    float* out = (float*)d_out;

    const int* src = eidx;
    const int* dst = eidx + N_EDGES;

    // bump allocator on d_ws
    char* base = (char*)d_ws;
    size_t off = 0;
    auto alloc = [&](size_t bytes) -> char* {
        char* p = base + off;
        off = (off + bytes + 255) & ~(size_t)255;
        return p;
    };
    float* inv_deg = (float*)alloc((size_t)N_NODES * 4);
    float* h       = (float*)alloc((size_t)N_NODES * HID * 4);
    float* agg     = (float*)alloc((size_t)N_NODES * HID * 4);     // also GEMM out
    h16*   a16     = (h16*)  alloc((size_t)N_NODES * HID * 2);
    h16*   multi16 = (h16*)  alloc((size_t)N_NODES * MULTI * 2);   // concat reps, f16
    h16*   fush16  = (h16*)  alloc((size_t)N_NODES * HID * 2);
    h16*   x16     = (h16*)  alloc((size_t)N_NODES * IN_PAD * 2);
    h16*   embWt   = (h16*)  alloc((size_t)HID * IN_PAD * 2);
    h16*   linlWt  = (h16*)  alloc((size_t)N_LAYERS * HID * HID * 2);
    h16*   linrWt  = (h16*)  alloc((size_t)N_LAYERS * HID * HID * 2);
    h16*   fus1Wt  = (h16*)  alloc((size_t)HID * MULTI * 2);
    h16*   fus2Wt  = (h16*)  alloc((size_t)HID * HID * 2);
    (void)ws_size; (void)in_sizes; (void)n_in; (void)out_size;

    const int TB = 256;
    auto blocks = [&](long n) { return (int)((n + TB - 1) / TB); };

    // ---- degree / inverse degree ----
    k_fill_f32<<<blocks(N_NODES), TB, 0, stream>>>(inv_deg, N_NODES, 0.0f);
    k_deg<<<blocks(N_EDGES), TB, 0, stream>>>(dst, inv_deg);
    k_invdeg<<<blocks(N_NODES), TB, 0, stream>>>(inv_deg);

    // ---- convert inputs / weights to f16 (weights transposed to [N][K]) ----
    k_cvt_x<<<blocks((long)N_NODES * IN_PAD), TB, 0, stream>>>(x, x16);
    k_wt_cvt<<<blocks((long)HID * IN_PAD), TB, 0, stream>>>(emb_W, embWt, IN_DIM, IN_PAD);
    for (int i = 0; i < N_LAYERS; ++i) {
        k_wt_cvt<<<blocks((long)HID * HID), TB, 0, stream>>>(
            lin_l_W + (size_t)i * HID * HID, linlWt + (size_t)i * HID * HID, HID, HID);
        k_wt_cvt<<<blocks((long)HID * HID), TB, 0, stream>>>(
            lin_r_W + (size_t)i * HID * HID, linrWt + (size_t)i * HID * HID, HID, HID);
    }
    k_wt_cvt<<<blocks((long)HID * MULTI), TB, 0, stream>>>(fus_W1, fus1Wt, MULTI, MULTI);
    k_wt_cvt<<<blocks((long)HID * HID), TB, 0, stream>>>(fus_W2, fus2Wt, HID, HID);

    dim3 gblk(32, 8);
    int  mtiles = (N_NODES + 15) / 16;
    dim3 lblk(32, 8);
    int  lgrid = (N_NODES + 7) / 8;

    // ---- embedding: h = relu(x @ emb_W + emb_b); rep0 = f16(h) ----
    k_gemm<<<mtiles, gblk, 0, stream>>>(x16, IN_PAD, embWt, IN_PAD,
                                        (const h16*)nullptr, 0, (const h16*)nullptr, 0,
                                        emb_b, h, multi16 /*block 0*/, MULTI,
                                        N_NODES, 1);

    // ---- SAGE layers ----
    for (int i = 0; i < N_LAYERS; ++i) {
        k_fill_f32<<<blocks((long)N_NODES * HID), TB, 0, stream>>>(agg, N_NODES * HID, 0.0f);
        k_scatter<<<blocks((long)N_EDGES * (HID / 4)), TB, 0, stream>>>(src, dst, h, agg);
        k_scale_cvt<<<blocks((long)N_NODES * HID), TB, 0, stream>>>(agg, inv_deg, a16);

        // agg(out f32) = a16 @ lin_l[i]^T + b_l[i] + rep_i @ lin_r[i]^T
        k_gemm<<<mtiles, gblk, 0, stream>>>(
            a16, HID, linlWt + (size_t)i * HID * HID, HID,
            multi16 + (size_t)i * HID, MULTI, linrWt + (size_t)i * HID * HID, HID,
            lin_l_b + (size_t)i * HID,
            agg, (h16*)nullptr, 0, N_NODES, 0);

        // h = relu(LN(agg) [+ h if i>0]); rep_{i+1} = f16(h)
        k_ln<<<lgrid, lblk, 0, stream>>>(agg, h, multi16 + (size_t)(i + 1) * HID, MULTI,
                                         ln_g + (size_t)i * HID, ln_b + (size_t)i * HID,
                                         (i > 0) ? 1 : 0);
    }

    // ---- fusion MLP ----
    // fush16 = f16(relu(multi @ fus_W1 + b1))
    k_gemm<<<mtiles, gblk, 0, stream>>>(multi16, MULTI, fus1Wt, MULTI,
                                        (const h16*)nullptr, 0, (const h16*)nullptr, 0,
                                        fus_b1, (float*)nullptr, fush16, HID,
                                        N_NODES, 1);
    // out = fush16 @ fus_W2 + b2
    k_gemm<<<mtiles, gblk, 0, stream>>>(fush16, HID, fus2Wt, HID,
                                        (const h16*)nullptr, 0, (const h16*)nullptr, 0,
                                        fus_b2, out, (h16*)nullptr, 0,
                                        N_NODES, 0);
}
